// RNN_30159260352544
// MI455X (gfx1250) — compile-verified, matrix operand order checked by reading, not run on hardware
//
#include <hip/hip_runtime.h>
#include <hip/hip_bf16.h>

// Problem constants (from reference): B=64, S=2048, IN=256, H=512
#define B_   64
#define S_   2048
#define IN_  256
#define H_   512
#define NWG  16          // persistent workgroups for the recurrence scan

typedef __attribute__((ext_vector_type(16))) __bf16 v16bf;
typedef __attribute__((ext_vector_type(8)))  float  v8f;
typedef __attribute__((ext_vector_type(4)))  int    v4i;

// ---------- CDNA5 async global->LDS copy (ASYNCcnt-tracked DMA) ----------
#if defined(__AMDGCN__) && __has_builtin(__builtin_amdgcn_global_load_async_to_lds_b128)
#define HAVE_ASYNC_LDS 1
#else
#define HAVE_ASYNC_LDS 0
#endif

typedef __attribute__((address_space(1))) v4i gv4i;  // global int4
typedef __attribute__((address_space(3))) v4i sv4i;  // LDS int4

__device__ __forceinline__ void ld_async_b128(uint4* lds_dst, const uint4* g_src) {
#if HAVE_ASYNC_LDS
  __builtin_amdgcn_global_load_async_to_lds_b128(
      (gv4i*)(g_src), (sv4i*)(lds_dst), /*imm offset=*/0, /*cpol=*/0);
#else
  *lds_dst = *g_src;
#endif
}

__device__ __forceinline__ void wait_async_all() {
#if HAVE_ASYNC_LDS
#if __has_builtin(__builtin_amdgcn_s_wait_asynccnt)
  __builtin_amdgcn_s_wait_asynccnt(0);
#else
  asm volatile("s_wait_asynccnt 0x0" ::: "memory");
#endif
#endif
}

// ---------- helpers ----------

__device__ __forceinline__ __bf16 f2bf(float f) {
  // round-to-nearest-even f32 -> bf16
  unsigned u = __builtin_bit_cast(unsigned, f);
  unsigned lsb = (u >> 16) & 1u;
  u += 0x7fffu + lsb;
  return __builtin_bit_cast(__bf16, (unsigned short)(u >> 16));
}

__device__ __forceinline__ v8f wmma_bf16(v16bf a, v16bf b, v8f c) {
  // D = A(16x32 bf16) * B(32x16 bf16) + C(16x16 f32)
  return __builtin_amdgcn_wmma_f32_16x16x32_bf16(
      /*neg_a=*/false, a, /*neg_b=*/false, b,
      /*c_mod=*/(short)0, c, /*reuse_a=*/false, /*reuse_b=*/false);
}

// Grid-wide barrier: monotone generation counter in global memory.
// ctr is zeroed by hipMemsetAsync before the kernel launches.
__device__ __forceinline__ void grid_barrier(unsigned* ctr, unsigned gen) {
  __threadfence();
  __syncthreads();
  if (threadIdx.x == 0) {
    __hip_atomic_fetch_add(ctr, 1u, __ATOMIC_RELEASE, __HIP_MEMORY_SCOPE_AGENT);
    unsigned target = gen * NWG;
    while (__hip_atomic_load(ctr, __ATOMIC_ACQUIRE, __HIP_MEMORY_SCOPE_AGENT) < target) {
      __builtin_amdgcn_s_sleep(1);
    }
  }
  // NOP when not dispatched as a cluster (ClusterID==0); free when it is.
  __builtin_amdgcn_s_cluster_barrier();
  __syncthreads();
  __threadfence();
}

// ---------- pack f32 row-major [R,K] into bf16 WMMA fragment layout ----------
// Fragment layout per 16(M/N) x 32(K) tile (ISA 7.12.2, 16-bit A-matrix):
//   lane<16 : row = lane,    K offsets {0..7, 16..23}
//   lane>=16: row = lane-16, K offsets {8..15, 24..31}
// dst layout: [rtile][ktile][lane][16] bf16, each lane's 16 values contiguous.
__global__ void pack_frag_kernel(const float* __restrict__ src,
                                 __bf16* __restrict__ dst, int R, int K) {
  int t    = blockIdx.x * blockDim.x + threadIdx.x;
  int lane = t & 31;
  int tile = t >> 5;
  int nkt  = K >> 5;
  int ntiles = (R >> 4) * nkt;
  if (tile >= ntiles) return;
  int rt = tile / nkt;
  int kt = tile - rt * nkt;
  int r  = rt * 16 + (lane & 15);
  int kb = kt * 32 + ((lane >> 4) << 3);
  const float* s = src + (size_t)r * K + kb;
  v16bf v;
#pragma unroll
  for (int j = 0; j < 8; ++j) {
    v[j]     = f2bf(s[j]);
    v[j + 8] = f2bf(s[16 + j]);
  }
  *reinterpret_cast<v16bf*>(dst + (size_t)(tile * 32 + lane) * 16) = v;
}

// ---------- Phase 1: xh = x @ w_xh^T  (M=131072, N=512, K=256) ----------
// Block: 256 threads = 8 waves. Block tile: 128(M) x 64(N).
// Wave w: M-tile (blockIdx.y*8 + w), computes 4 N-tiles; K loop = 8 WMMA steps.
__global__ __launch_bounds__(256) void xh_gemm_kernel(
    const __bf16* __restrict__ Ap,   // packed x      [8192 mtiles][8 kt][32][16]
    const __bf16* __restrict__ Bp,   // packed w_xh   [32 ntiles][8 kt][32][16]
    float* __restrict__ xh) {        // [M=131072, H=512] f32
  __shared__ __align__(32) __bf16 ldsB[4 * 8 * 32 * 16];  // 32 KB: this block's B tiles

  const int tid = threadIdx.x, lane = tid & 31, w = tid >> 5;
  const int nblk = blockIdx.x;   // 0..7  (64 cols each)
  const int mblk = blockIdx.y;   // 0..1023 (128 rows each)

  {  // async DMA stage of 4 N-tiles x full K into LDS (16384 bf16 = 2048 uint4)
    const uint4* s = reinterpret_cast<const uint4*>(Bp + (size_t)nblk * 4 * 8 * 32 * 16);
    uint4* d = reinterpret_cast<uint4*>(ldsB);
#pragma unroll
    for (int j = 0; j < 8; ++j) ld_async_b128(d + tid + 256 * j, s + tid + 256 * j);
    wait_async_all();
  }
  __syncthreads();

  v8f c0 = {}, c1 = {}, c2 = {}, c3 = {};
  const int mtile = mblk * 8 + w;
  const __bf16* aBase = Ap + (size_t)mtile * 8 * 32 * 16;
#pragma unroll
  for (int kt = 0; kt < 8; ++kt) {
    v16bf a  = *reinterpret_cast<const v16bf*>(aBase + (size_t)(kt * 32 + lane) * 16);
    v16bf b0 = *reinterpret_cast<const v16bf*>(ldsB + ((size_t)(0 * 8 + kt) * 32 + lane) * 16);
    v16bf b1 = *reinterpret_cast<const v16bf*>(ldsB + ((size_t)(1 * 8 + kt) * 32 + lane) * 16);
    v16bf b2 = *reinterpret_cast<const v16bf*>(ldsB + ((size_t)(2 * 8 + kt) * 32 + lane) * 16);
    v16bf b3 = *reinterpret_cast<const v16bf*>(ldsB + ((size_t)(3 * 8 + kt) * 32 + lane) * 16);
    c0 = wmma_bf16(a, b0, c0);
    c1 = wmma_bf16(a, b1, c1);
    c2 = wmma_bf16(a, b2, c2);
    c3 = wmma_bf16(a, b3, c3);
  }
  // D layout: VGPR r -> M = r (+8 for lanes>=16), N = lane&15
  const int rowB = mtile * 16 + ((lane >> 4) << 3);
  const int colB = nblk * 64 + (lane & 15);
#pragma unroll
  for (int r = 0; r < 8; ++r) {
    float* o = xh + (size_t)(rowB + r) * H_ + colB;
    o[0]  = c0[r];
    o[16] = c1[r];
    o[32] = c2[r];
    o[48] = c3[r];
  }
}

// ---------- Phase 2: persistent recurrence scan ----------
// 16 WGs x 256 threads. WG i owns h cols [32i,32i+32) and out cols [16i,16i+16).
// h(t) exchanged via double-buffered bf16 fragment buffers in global (L2-resident).
__global__ __launch_bounds__(256) void rnn_scan_kernel(
    const float* __restrict__ xh,       // [B*S, H] f32
    const __bf16* __restrict__ whh_p,   // packed w_hh [32 nt][16 kt][32][16]
    const __bf16* __restrict__ who_p,   // packed w_ho [16 nt][16 kt][32][16]
    const float* __restrict__ b_hh,     // [H]
    const float* __restrict__ b_ho,     // [IN]
    const float* __restrict__ h0,       // [H]
    float* __restrict__ outs,           // [B,S,IN]
    float* __restrict__ hs,             // [B,S,H]
    __bf16* __restrict__ hbuf0,         // h frag buffer A: [4 mt][16 kt][32][16]
    __bf16* __restrict__ hbuf1,         // h frag buffer B
    unsigned* __restrict__ ctr) {
  __shared__ __align__(32) __bf16 lds_whh[2 * 16 * 32 * 16];  // 32 KB (our 2 N-tiles, full K)
  __shared__ __align__(32) __bf16 lds_who[16 * 32 * 16];      // 16 KB (our 1 N-tile, full K)
  __shared__ __align__(16) float  lds_h[4 * 16 * 32];         //  8 KB: h(t) slice for repack
  __shared__ float lds_bhh[32];
  __shared__ float lds_bho[16];

  const int wg = blockIdx.x;            // 0..15
  const int tid = threadIdx.x, lane = tid & 31, w = tid >> 5;

  {  // async DMA: pin weight slices in LDS for the whole scan
    const uint4* s1 = reinterpret_cast<const uint4*>(whh_p + (size_t)(wg * 2) * 16 * 32 * 16);
    uint4* d1 = reinterpret_cast<uint4*>(lds_whh);
#pragma unroll
    for (int j = 0; j < 8; ++j) ld_async_b128(d1 + tid + 256 * j, s1 + tid + 256 * j);
    const uint4* s2 = reinterpret_cast<const uint4*>(who_p + (size_t)wg * 16 * 32 * 16);
    uint4* d2 = reinterpret_cast<uint4*>(lds_who);
#pragma unroll
    for (int j = 0; j < 4; ++j) ld_async_b128(d2 + tid + 256 * j, s2 + tid + 256 * j);
    wait_async_all();
  }
  if (tid < 32) lds_bhh[tid] = b_hh[wg * 32 + tid];
  if (tid < 16) lds_bho[tid] = b_ho[wg * 16 + tid];

  // h(-1) = broadcast(h0): pack our ktile (=wg) of the A-fragment buffer
  if (tid < 128) {
    int mt = tid >> 5, l = tid & 31;
    int kb = wg * 32 + ((l >> 4) << 3);
    v16bf v;
#pragma unroll
    for (int j = 0; j < 8; ++j) {
      v[j]     = f2bf(h0[kb + j]);
      v[j + 8] = f2bf(h0[kb + 16 + j]);
    }
    *reinterpret_cast<v16bf*>(hbuf0 + (size_t)((mt * 16 + wg) * 32 + l) * 16) = v;
  }
  __syncthreads();
  unsigned gen = 1;
  grid_barrier(ctr, gen++);  // h(-1) fragments visible device-wide

  const int mtile = w >> 1;  // GEMM1: wave -> (M-tile, local N-tile)
  const int ntl   = w & 1;

  for (int t = 0; t < S_; ++t) {
    const __bf16* prev = (t & 1) ? hbuf1 : hbuf0;
    __bf16* cur        = (t & 1) ? hbuf0 : hbuf1;

    // ---- GEMM1: h(t)[.,:32 slice] = tanh(xh_t + h(t-1) @ w_hh^T + b_hh) ----
    v8f c = {};
#pragma unroll 4
    for (int kt = 0; kt < 16; ++kt) {
      v16bf a = *reinterpret_cast<const v16bf*>(prev + (size_t)((mtile * 16 + kt) * 32 + lane) * 16);
      v16bf b = *reinterpret_cast<const v16bf*>(lds_whh + ((size_t)(ntl * 16 + kt) * 32 + lane) * 16);
      c = wmma_bf16(a, b, c);
    }
    {
      const int colL = ntl * 16 + (lane & 15);
      const int col  = wg * 32 + colL;
      const int rowB = mtile * 16 + ((lane >> 4) << 3);
      const float bias = lds_bhh[colL];
      if (t + 1 < S_)  // global_prefetch_b8 for next step's xh lines
        __builtin_prefetch(xh + ((size_t)rowB * S_ + (t + 1)) * H_ + col, 0, 1);
#pragma unroll
      for (int r = 0; r < 8; ++r) {
        const int row = rowB + r;  // batch index b
        float v = tanhf(c[r] + xh[((size_t)row * S_ + t) * H_ + col] + bias);
        hs[((size_t)row * S_ + t) * H_ + col] = v;
        lds_h[(size_t)(mtile * 16 + (row - mtile * 16)) * 32 + colL] = v;
      }
    }
    __syncthreads();

    // ---- repack h(t) slice -> bf16 A-fragments (our ktile = wg) ----
    if (tid < 128) {
      int mt = tid >> 5, l = tid & 31;
      int kb = (l >> 4) << 3;
      const float* hsrc = lds_h + (size_t)(mt * 16 + (l & 15)) * 32;
      v16bf v;
#pragma unroll
      for (int j = 0; j < 8; ++j) {
        v[j]     = f2bf(hsrc[kb + j]);
        v[j + 8] = f2bf(hsrc[kb + 16 + j]);
      }
      *reinterpret_cast<v16bf*>(cur + (size_t)((mt * 16 + wg) * 32 + l) * 16) = v;
    }
    grid_barrier(ctr, gen++);  // full h(t) fragment buffer visible

    // ---- GEMM2: out(t)[.,:16 slice] = tanh(h(t) @ w_ho^T + b_ho) ----
    if (w < 4) {
      v8f c2 = {};
#pragma unroll 4
      for (int kt = 0; kt < 16; ++kt) {
        v16bf a = *reinterpret_cast<const v16bf*>(cur + (size_t)((w * 16 + kt) * 32 + lane) * 16);
        v16bf b = *reinterpret_cast<const v16bf*>(lds_who + ((size_t)kt * 32 + lane) * 16);
        c2 = wmma_bf16(a, b, c2);
      }
      const int col  = wg * 16 + (lane & 15);
      const int rowB = w * 16 + ((lane >> 4) << 3);
      const float bias = lds_bho[lane & 15];
#pragma unroll
      for (int r = 0; r < 8; ++r) {
        const int row = rowB + r;
        outs[((size_t)row * S_ + t) * IN_ + col] = tanhf(c2[r] + bias);
      }
    }
  }
}

// ---------- launcher ----------
extern "C" void kernel_launch(void* const* d_in, const int* in_sizes, int n_in,
                              void* d_out, int out_size, void* d_ws, size_t ws_size,
                              hipStream_t stream) {
  (void)in_sizes; (void)n_in; (void)out_size; (void)ws_size;
  const float* x    = (const float*)d_in[0];  // [B,S,IN]
  const float* w_xh = (const float*)d_in[1];  // [H,IN]
  const float* w_hh = (const float*)d_in[2];  // [H,H]
  const float* b_hh = (const float*)d_in[3];  // [H]
  const float* w_ho = (const float*)d_in[4];  // [IN,H]
  const float* b_ho = (const float*)d_in[5];  // [IN]
  const float* h0   = (const float*)d_in[6];  // [1,H]

  float* outs = (float*)d_out;                          // [B,S,IN]
  float* hs   = outs + (size_t)B_ * S_ * IN_;           // [B,S,H]

  // workspace carve-up (all offsets 32B-aligned)
  char* ws = (char*)d_ws;
  unsigned* ctr  = (unsigned*)ws;                                   // 256 B
  __bf16* hbuf0  = (__bf16*)(ws + 256);                             // 64 KB
  __bf16* hbuf1  = (__bf16*)(ws + 256 + 65536);                     // 64 KB
  __bf16* wxh_p  = (__bf16*)(ws + 256 + 131072);                    // 256 KB
  __bf16* whh_p  = wxh_p + (size_t)H_ * IN_;                        // 512 KB
  __bf16* who_p  = whh_p + (size_t)H_ * H_;                         // 256 KB
  __bf16* x_p    = who_p + (size_t)IN_ * H_;                        // 64 MB
  float*  xh     = (float*)(x_p + (size_t)B_ * S_ * IN_);           // 256 MB

  (void)hipMemsetAsync(ctr, 0, 256, stream);  // grid-barrier generation counter

  const int M = B_ * S_;  // 131072
  // fragment packing: x, w_xh, w_hh, w_ho
  pack_frag_kernel<<<(((M / 16) * (IN_ / 32) * 32) + 255) / 256, 256, 0, stream>>>(x, x_p, M, IN_);
  pack_frag_kernel<<<(((H_ / 16) * (IN_ / 32) * 32) + 255) / 256, 256, 0, stream>>>(w_xh, wxh_p, H_, IN_);
  pack_frag_kernel<<<(((H_ / 16) * (H_ / 32) * 32) + 255) / 256, 256, 0, stream>>>(w_hh, whh_p, H_, H_);
  pack_frag_kernel<<<(((IN_ / 16) * (H_ / 32) * 32) + 255) / 256, 256, 0, stream>>>(w_ho, who_p, IN_, H_);

  // Phase 1: xh = x @ w_xh^T
  xh_gemm_kernel<<<dim3(H_ / 64, M / 128), 256, 0, stream>>>(x_p, wxh_p, xh);

  // Phase 2: persistent recurrence scan
  rnn_scan_kernel<<<NWG, 256, 0, stream>>>(xh, whh_p, who_p, b_hh, b_ho, h0,
                                           outs, hs, hbuf0, hbuf1, ctr);
}
